// QueryEmb_70935679861253
// MI455X (gfx1250) — matrix operand order repaired and checked by reading, not compiled, not run on hardware
//
#include <hip/hip_runtime.h>

// ---------------------------------------------------------------------------
// QueryEmb segment-softmax attention for MI455X (gfx1250, wave32, WMMA).
//   out[b,l,:] = softmax_{v in seg(x_rank)}(q . emb_v / sqrt(512))^T @ emb + q
// Bucket tokens by rank -> two-pass bf16 WMMA attention (K=V=emb segment).
// Pass2: 8-wave cooperative workgroups, double-buffered shared P tile,
// v-split x2 with linear partial-ctx combine for full-machine occupancy.
// ---------------------------------------------------------------------------

#define BTOK   2048          // B*L
#define DMODEL 512
#define VOCAB  32000
#define NSEG   4
#define SEGV   8000          // VOCAB/NSEG
#define NCHUNK 500           // SEGV/16
#define VSPLIT 8             // pass1 vocab splits
#define CSPLIT 2             // pass2 vocab splits (250 chunks each; stays even)
#define ROWS_PER_BUCKET 2048
#define TOTROWS (NSEG*ROWS_PER_BUCKET)   // 8192
#define INV_TEMP 0.04419417382415922f    // 1/sqrt(512)

typedef unsigned short u16;
typedef __attribute__((ext_vector_type(16))) __bf16 v16bf;
typedef __attribute__((ext_vector_type(8)))  float  v8f;

union FragU { uint4 u[2]; v16bf v; };

__device__ __forceinline__ u16 f2bf(float x) {
    unsigned int u = __float_as_uint(x);
    u = (u + 0x7fffu + ((u >> 16) & 1u)) >> 16;
    return (u16)u;
}

__device__ __forceinline__ v8f wmma_bf16(v16bf a, v16bf b, v8f c) {
    return __builtin_amdgcn_wmma_f32_16x16x32_bf16(false, a, false, b,
                                                   (short)0, c, false, false);
}

// A-matrix 16x32 bf16 fragment (lane = M; halves hold K 0-7/16-23 vs 8-15/24-31)
__device__ __forceinline__ v16bf load_a_frag(const u16* row0, int stride, int kc, int lane) {
    int m = lane & 15, h = (lane >> 4) & 1;
    const u16* p = row0 + (size_t)m * stride;
    FragU f;
    f.u[0] = *(const uint4*)(p + kc + h * 8);
    f.u[1] = *(const uint4*)(p + kc + 16 + h * 8);
    return f.v;
}

// B-matrix 32x16 bf16 fragment, B[k][n] = mat[v0+n][kc+k] (lane = N; halves hold K 0-7/16-23 vs 8-15/24-31)
__device__ __forceinline__ v16bf load_b_frag(const u16* mat, int v0, int stride, int kc, int lane) {
    int n = lane & 15, h = (lane >> 4) & 1;
    const u16* p = mat + (size_t)(v0 + n) * stride + kc + h * 8;
    FragU f;
    f.u[0] = *(const uint4*)(p);
    f.u[1] = *(const uint4*)(p + 16);
    return f.v;
}

// ---------------------------------------------------------------------------
// One-time: emb f32 -> bf16 row-major (Ebf) AND bf16 transposed (ETbf).
__global__ __launch_bounds__(256) void prep_emb(const float* __restrict__ e,
                                                u16* __restrict__ ebf,
                                                u16* __restrict__ etbf) {
    __shared__ u16 t[32][33];
    int v0 = blockIdx.x * 32, d0 = blockIdx.y * 32;
    int tx = threadIdx.x & 31, ty = threadIdx.x >> 5;
#pragma unroll
    for (int rr = 0; rr < 4; ++rr) {
        int i = ty + rr * 8;
        u16 b = f2bf(e[(size_t)(v0 + i) * DMODEL + d0 + tx]);
        ebf[(size_t)(v0 + i) * DMODEL + d0 + tx] = b;
        t[i][tx] = b;
    }
    __syncthreads();
#pragma unroll
    for (int rr = 0; rr < 4; ++rr) {
        int i = ty + rr * 8;
        etbf[(size_t)(d0 + i) * VOCAB + v0 + tx] = t[tx][i];
    }
}

__global__ __launch_bounds__(1024) void bucketize(const int* __restrict__ xr,
                                                  int* __restrict__ rowmap,
                                                  int* __restrict__ counts) {
    __shared__ int lcnt[NSEG];
    int tid = threadIdx.x;
    if (tid < NSEG) lcnt[tid] = 0;
    for (int i = tid; i < TOTROWS; i += 1024) rowmap[i] = -1;
    __syncthreads();
    for (int t = tid; t < BTOK; t += 1024) {
        int rk = xr[t] & 3;
        int pos = atomicAdd(&lcnt[rk], 1);
        rowmap[rk * ROWS_PER_BUCKET + pos] = t;
    }
    __syncthreads();
    if (tid < NSEG) counts[tid] = lcnt[tid];
}

__global__ void gather_q(const float* __restrict__ q, const int* __restrict__ rowmap,
                         const int* __restrict__ counts, u16* __restrict__ qb) {
    int row = blockIdx.x;
    int tid = threadIdx.x;                      // 64 threads
    int tok = rowmap[row];
    int r = row >> 11, rr = row & (ROWS_PER_BUCKET - 1);
    int nr = (counts[r] + 15) & ~15;
    if (tok >= 0) {
        for (int j = tid; j < DMODEL; j += 64)
            qb[(size_t)row * DMODEL + j] = f2bf(q[(size_t)tok * DMODEL + j]);
    } else if (rr < nr) {
        for (int j = tid; j < DMODEL; j += 64)
            qb[(size_t)row * DMODEL + j] = 0;   // zero pad rows
    }
}

// ---------------------------------------------------------------------------
// Pass 1: per 16-token tile, per v-split: running (max, sum) over the segment.
__global__ __launch_bounds__(32) void pass1(const u16* __restrict__ ebf,
                                            const u16* __restrict__ qb,
                                            const int* __restrict__ counts,
                                            float* __restrict__ pml) {
    int tile = blockIdx.x, r = blockIdx.y, vs = blockIdx.z;
    int n = counts[r];
    if (tile * 16 >= n) return;                  // uniform: EXEC stays all-1
    int lane = threadIdx.x;
    int half = lane >> 4;
    int rowbase = r * ROWS_PER_BUCKET + tile * 16;

    v16bf qa[16];
#pragma unroll
    for (int kf = 0; kf < 16; ++kf)
        qa[kf] = load_a_frag(qb + (size_t)rowbase * DMODEL, DMODEL, kf * 32, lane);

    float rm[8], rl[8];
#pragma unroll
    for (int i = 0; i < 8; ++i) { rm[i] = -3.4e38f; rl[i] = 0.f; }

    int c0 = vs * 63;
    int c1 = c0 + 63; if (c1 > NCHUNK) c1 = NCHUNK;
    for (int c = c0; c < c1; ++c) {
        int v0 = r * SEGV + c * 16;
        __builtin_prefetch(ebf + (size_t)(v0 + 16 + (lane & 15)) * DMODEL + half * 256, 0, 3);
        v8f acc = {0.f, 0.f, 0.f, 0.f, 0.f, 0.f, 0.f, 0.f};
#pragma unroll
        for (int kf = 0; kf < 16; ++kf) {
            v16bf b = load_b_frag(ebf, v0, DMODEL, kf * 32, lane);
            acc = wmma_bf16(qa[kf], b, acc);
        }
#pragma unroll
        for (int i = 0; i < 8; ++i) {
            float s = acc[i] * INV_TEMP;
            float mx = s;
            for (int d = 1; d < 16; d <<= 1) mx = fmaxf(mx, __shfl_xor(mx, d, 32));
            float newm = fmaxf(rm[i], mx);
            float e = __expf(s - newm);
            float se = e;
            for (int d = 1; d < 16; d <<= 1) se += __shfl_xor(se, d, 32);
            rl[i] = rl[i] * __expf(rm[i] - newm) + se;
            rm[i] = newm;
        }
    }
    if ((lane & 15) == 0) {
#pragma unroll
        for (int i = 0; i < 8; ++i) {
            int row = rowbase + i + half * 8;
            int idx = (row * VSPLIT + vs) * 2;
            pml[idx] = rm[i];
            pml[idx + 1] = rl[i];
        }
    }
}

__global__ void merge_ml(const int* __restrict__ counts, const float* __restrict__ pml,
                         float* __restrict__ ml) {
    int row = blockIdx.x * blockDim.x + threadIdx.x;
    if (row >= TOTROWS) return;
    int r = row >> 11, rr = row & (ROWS_PER_BUCKET - 1);
    int nr = (counts[r] + 15) & ~15;
    if (rr >= nr) { ml[row * 2] = 0.f; ml[row * 2 + 1] = 1.f; return; }
    float M = -3.4e38f;
    for (int s = 0; s < VSPLIT; ++s) M = fmaxf(M, pml[(row * VSPLIT + s) * 2]);
    float L = 0.f;
    for (int s = 0; s < VSPLIT; ++s)
        L += pml[(row * VSPLIT + s) * 2 + 1] * __expf(pml[(row * VSPLIT + s) * 2] - M);
    ml[row * 2] = M;
    ml[row * 2 + 1] = L;
}

// ---------------------------------------------------------------------------
// Pass 2: cooperative 8-wave workgroup per (16-token tile, half-segment).
// Per iteration (128 vocab): wave w computes one 16x16 score tile (16 WMMA),
// drops normalized p into a double-buffered shared 16x128 P tile; after one
// barrier every wave runs ctx += P @ E for its own 64 columns (16 WMMA,
// contiguous B-frags from the transposed emb). Partial ctx goes to ws f32.
__global__ __launch_bounds__(256) void pass2(const u16* __restrict__ ebf,
                                             const u16* __restrict__ etbf,
                                             const u16* __restrict__ qb,
                                             const int* __restrict__ counts,
                                             const float* __restrict__ ml,
                                             float* __restrict__ cpart) {
    __shared__ u16 plds[2][16 * 128];  // double-buffered P tile [m][kk]
    int tile = blockIdx.x, r = blockIdx.y, vs = blockIdx.z;
    int n = counts[r];
    if (tile * 16 >= n) return;        // n uniform across block
    int tid = threadIdx.x;
    int w = tid >> 5, lane = tid & 31;
    int m15 = lane & 15, half = lane >> 4;
    int rowbase = r * ROWS_PER_BUCKET + tile * 16;
    int segbase = r * SEGV + vs * (SEGV / CSPLIT);
    int col0 = w * 64;

    v16bf qa[16];
#pragma unroll
    for (int kf = 0; kf < 16; ++kf)
        qa[kf] = load_a_frag(qb + (size_t)rowbase * DMODEL, DMODEL, kf * 32, lane);

    float Mrow[8], recL[8];
#pragma unroll
    for (int i = 0; i < 8; ++i) {
        int row = rowbase + i + half * 8;
        Mrow[i] = ml[row * 2];
        recL[i] = 1.0f / ml[row * 2 + 1];
    }

    v8f accC[4];
#pragma unroll
    for (int t = 0; t < 4; ++t) accC[t] = (v8f){0.f, 0.f, 0.f, 0.f, 0.f, 0.f, 0.f, 0.f};

    const int myChunks = NCHUNK / CSPLIT;          // 250 (even)
    const int nIter = (myChunks + 7) / 8;          // 32: 31 full + one of 2 chunks
    for (int it = 0; it < nIter; ++it) {
        int nch = myChunks - it * 8; if (nch > 8) nch = 8;
        u16* pbuf = plds[it & 1];
        if (w < nch) {
            int v0 = segbase + (it * 8 + w) * 16;
            __builtin_prefetch(ebf + (size_t)(v0 + 128 + m15) * DMODEL + half * 256, 0, 3);
            v8f s = {0.f, 0.f, 0.f, 0.f, 0.f, 0.f, 0.f, 0.f};
#pragma unroll
            for (int kf = 0; kf < 16; ++kf)
                s = wmma_bf16(qa[kf], load_b_frag(ebf, v0, DMODEL, kf * 32, lane), s);
#pragma unroll
            for (int i = 0; i < 8; ++i) {
                float p = __expf(s[i] * INV_TEMP - Mrow[i]) * recL[i];
                pbuf[(i + half * 8) * 128 + w * 16 + m15] = f2bf(p);
            }
        }
        __syncthreads();                           // single barrier (double buffer)
        int vb = segbase + it * 128;
        int nkc = nch >> 1;                        // K-chunks of 32 in this block
        for (int kc = 0; kc < nkc; ++kc) {
            FragU fa;
            const u16* pa = pbuf + m15 * 128 + kc * 32;
            fa.u[0] = *(const uint4*)(pa + half * 8);
            fa.u[1] = *(const uint4*)(pa + 16 + half * 8);
#pragma unroll
            for (int nt = 0; nt < 4; ++nt) {
                const u16* pb = etbf + (size_t)(col0 + nt * 16 + m15) * VOCAB
                              + vb + kc * 32 + half * 8;
                if (kc == 0)
                    __builtin_prefetch(pb + 128, 0, 3);   // next block's E^T run
                FragU fb;
                fb.u[0] = *(const uint4*)(pb);
                fb.u[1] = *(const uint4*)(pb + 16);
                accC[nt] = wmma_bf16(fa.v, fb.v, accC[nt]);
            }
        }
    }

    // --- write partial ctx tile (f32) for this v-split ---
    float* cp = cpart + (size_t)vs * TOTROWS * DMODEL;
#pragma unroll
    for (int i = 0; i < 8; ++i) {
        int mrow = i + half * 8;
#pragma unroll
        for (int nt = 0; nt < 4; ++nt)
            cp[(size_t)(rowbase + mrow) * DMODEL + col0 + nt * 16 + m15] = accC[nt][i];
    }
}

// Combine the CSPLIT partial ctx sums, add residual, scatter to token order.
__global__ __launch_bounds__(256) void combine_out(const float* __restrict__ q,
                                                   const float* __restrict__ cpart,
                                                   const int* __restrict__ rowmap,
                                                   float* __restrict__ out) {
    int row = blockIdx.x;
    int tok = rowmap[row];
    if (tok < 0) return;
    int tid = threadIdx.x;
    for (int j = tid; j < DMODEL; j += 256) {
        float v = cpart[(size_t)row * DMODEL + j]
                + cpart[(size_t)TOTROWS * DMODEL + (size_t)row * DMODEL + j];
        out[(size_t)tok * DMODEL + j] = v + q[(size_t)tok * DMODEL + j];
    }
}

// ---------------------------------------------------------------------------
extern "C" void kernel_launch(void* const* d_in, const int* in_sizes, int n_in,
                              void* d_out, int out_size, void* d_ws, size_t ws_size,
                              hipStream_t stream) {
    (void)in_sizes; (void)n_in; (void)out_size; (void)ws_size;
    const float* q   = (const float*)d_in[0];
    const int*   xr  = (const int*)d_in[1];
    const float* emb = (const float*)d_in[2];
    float* out = (float*)d_out;
    char* ws = (char*)d_ws;

    u16*   ebf    = (u16*)(ws + 0);                  // 32,768,000 B
    u16*   etbf   = (u16*)(ws + 32768000);           // 32,768,000 B (emb^T)
    u16*   qb     = (u16*)(ws + 65536000);           //  8,388,608 B
    int*   rowmap = (int*)(ws + 73924608);           //     32,768 B
    int*   counts = (int*)(ws + 73957376);           //        256 B
    float* pml    = (float*)(ws + 73957632);         //    524,288 B
    float* mlbuf  = (float*)(ws + 74481920);         //     65,536 B
    float* cpart  = (float*)(ws + 74547456);         // 33,554,432 B (2 splits)

    prep_emb<<<dim3(VOCAB / 32, DMODEL / 32), 256, 0, stream>>>(emb, ebf, etbf);
    bucketize<<<1, 1024, 0, stream>>>(xr, rowmap, counts);
    gather_q<<<TOTROWS, 64, 0, stream>>>(q, rowmap, counts, qb);
    pass1<<<dim3(ROWS_PER_BUCKET / 16, NSEG, VSPLIT), 32, 0, stream>>>(ebf, qb, counts, pml);
    merge_ml<<<TOTROWS / 256, 256, 0, stream>>>(counts, pml, mlbuf);
    pass2<<<dim3(ROWS_PER_BUCKET / 16, NSEG, CSPLIT), 256, 0, stream>>>(
        ebf, etbf, qb, counts, mlbuf, cpart);
    combine_out<<<TOTROWS, 256, 0, stream>>>(q, cpart, rowmap, out);
}
